// BitwiseLinear_6743098655627
// MI455X (gfx1250) — 8-bit matrix operand order repaired and checked by reading, compiled, not run on hardware
//
#include <hip/hip_runtime.h>

typedef __attribute__((ext_vector_type(8))) int   v8i;
typedef __attribute__((ext_vector_type(4))) int   v4i;
typedef __attribute__((ext_vector_type(4))) float v4f;

#define THRESH 0.05f
#define KDIM 1024
#define KT (KDIM / 64)
#define THREADS 256

__device__ __forceinline__ int tq(float v) {
    return (v >= THRESH ? 1 : 0) - (v <= -THRESH ? 1 : 0);
}

template <bool BLAY>
__global__ __launch_bounds__(THREADS)
void quantize_swizzle(const float* __restrict__ src, signed char* __restrict__ dst) {
    const int idx  = blockIdx.x * THREADS + threadIdx.x;
    const int row  = idx >> 8;
    const int q    = idx & 255;
    const int kb   = q >> 4;
    const int q16  = q & 15;
    const int tile = row >> 4;
    const int l16  = row & 15;

    int half, v;
    if (BLAY) {
        half = (q16 >> 2) & 1;
        v    = (q16 & 3) | ((q16 >> 3) << 2);
    } else {
        half = (q16 >> 2) & 1;
        v    = (q16 & 3) | ((q16 >> 3) << 2);
    }
    const int lane = l16 + (half << 4);

    v4f val = __builtin_nontemporal_load((const v4f*)(src) + idx);
    int packed = (tq(val[0]) & 0xFF)        | ((tq(val[1]) & 0xFF) << 8) |
                 ((tq(val[2]) & 0xFF) << 16) | ((tq(val[3]) & 0xFF) << 24);

    const int doff = ((tile * KT + kb) << 8) + (lane << 3) + v;
    ((int*)dst)[doff] = packed;
}

__device__ __forceinline__ v8i ldfrag(const signed char* p) {
    v4i lo = *(const v4i*)(p);
    v4i hi = *(const v4i*)(p + 16);
    v8i r;
    r[0] = lo[0]; r[1] = lo[1]; r[2] = lo[2]; r[3] = lo[3];
    r[4] = hi[0]; r[5] = hi[1]; r[6] = hi[2]; r[7] = hi[3];
    return r;
}

__global__ __launch_bounds__(THREADS)
void ternary_wmma_gemm(const signed char* __restrict__ xs,
                       const signed char* __restrict__ wsw,
                       float* __restrict__ out, int N) {
    const int t    = threadIdx.x;
    const int lane = t & 31;
    const int wave = t >> 5;
    const int hi16 = lane >> 4;
    const int l16  = lane & 15;

    const int tm0 = blockIdx.y * 8 + (wave & 3) * 2;
    const int tn0 = blockIdx.x * 4 + (wave >> 2) * 2;

    const signed char* pa0 = xs  + (size_t)(tm0 + 0) * (KT * 1024) + lane * 32;
    const signed char* pa1 = xs  + (size_t)(tm0 + 1) * (KT * 1024) + lane * 32;
    const signed char* pb0 = wsw + (size_t)(tn0 + 0) * (KT * 1024) + lane * 32;
    const signed char* pb1 = wsw + (size_t)(tn0 + 1) * (KT * 1024) + lane * 32;

    v8i acc[2][2] = {};

    #pragma unroll 4
    for (int kb = 0; kb < KT; ++kb) {
        const int off = kb * 1024;
        v8i a0 = ldfrag(pa0 + off);
        v8i a1 = ldfrag(pa1 + off);
        v8i b0 = ldfrag(pb0 + off);
        v8i b1 = ldfrag(pb1 + off);
        acc[0][0] = __builtin_amdgcn_wmma_i32_16x16x64_iu8(true, a0, true, b0, acc[0][0], false, false);
        acc[0][1] = __builtin_amdgcn_wmma_i32_16x16x64_iu8(true, a0, true, b1, acc[0][1], false, false);
        acc[1][0] = __builtin_amdgcn_wmma_i32_16x16x64_iu8(true, a1, true, b0, acc[1][0], false, false);
        acc[1][1] = __builtin_amdgcn_wmma_i32_16x16x64_iu8(true, a1, true, b1, acc[1][1], false, false);
    }

    #pragma unroll
    for (int mi = 0; mi < 2; ++mi) {
        #pragma unroll
        for (int ni = 0; ni < 2; ++ni) {
            const int col = (tn0 + ni) * 16 + l16;
            #pragma unroll
            for (int r = 0; r < 8; ++r) {
                const int row = (tm0 + mi) * 16 + r + hi16 * 8;
                __builtin_nontemporal_store((float)acc[mi][ni][r],
                                            &out[(size_t)row * N + col]);
            }
        }
    }
}

extern "C" void kernel_launch(void* const* d_in, const int* in_sizes, int n_in,
                              void* d_out, int out_size, void* d_ws, size_t ws_size,
                              hipStream_t stream) {
    (void)n_in; (void)out_size; (void)ws_size;
    const float* x = (const float*)d_in[0];
    const float* w = (const float*)d_in[1];
    float* out = (float*)d_out;

    const int K = KDIM;
    const int N = in_sizes[1] / K;
    const int M = in_sizes[0] / K;

    signed char* xs  = (signed char*)d_ws;
    size_t xBytes    = (size_t)M * K;
    signed char* wsw = (signed char*)d_ws + ((xBytes + 1023) & ~(size_t)1023);

    quantize_swizzle<false><<<(M * (K / 4)) / THREADS, THREADS, 0, stream>>>(x, xs);
    quantize_swizzle<true ><<<(N * (K / 4)) / THREADS, THREADS, 0, stream>>>(w, wsw);

    dim3 grid(N / 64, M / 128);
    ternary_wmma_gemm<<<grid, THREADS, 0, stream>>>(xs, wsw, out, N);
}
